// GNN_layer_51453708206082
// MI455X (gfx1250) — matrix operand order repaired and checked by reading, not compiled, not run on hardware
//
#include <hip/hip_runtime.h>
#include <hip/hip_bf16.h>

typedef __bf16 bf16_t;
typedef __attribute__((ext_vector_type(16))) __bf16 v16bf;
typedef __attribute__((ext_vector_type(8)))  __bf16 v8bf;
typedef __attribute__((ext_vector_type(8)))  float   v8f;

union FragBF { v16bf v; v8bf h[2]; };

// ---------------------------------------------------------------------------
// Fragment loaders (CDNA5 wave32 WMMA layouts, CDNA5 ISA §7.12.2)
// A (16x32 bf16): lane L -> M = L%16; elems 0..7 = K k0+laneHi*8+0..7,
//                 elems 8..15 = K k0+16+laneHi*8+0..7   (two 16B LDS loads)
// B (32x16 bf16): lane L -> N = L%16; elem e -> K = k0 + 16*(e/8) + laneHi*8 + e%8
//                 (pre-packed contiguous: one 32B global load)
// C/D (16x16 f32): lane -> N = lane%16; vgpr r -> M = r + 8*laneHi
// ---------------------------------------------------------------------------
static __device__ __forceinline__ v16bf load_a_frag(const bf16_t* base, int stride,
                                                    int m0, int k0, int lane) {
  const bf16_t* p = base + (m0 + (lane & 15)) * stride + k0 + ((lane >> 4) << 3);
  FragBF f;
  f.h[0] = *(const v8bf*)(p);
  f.h[1] = *(const v8bf*)(p + 16);
  return f.v;
}

static __device__ __forceinline__ v16bf load_b_frag(const bf16_t* wpk, int fragIdx, int lane) {
  const bf16_t* p = wpk + ((size_t)fragIdx << 9) + (lane << 4);
  FragBF f;
  f.h[0] = *(const v8bf*)(p);
  f.h[1] = *(const v8bf*)(p + 8);
  return f.v;
}

static __device__ __forceinline__ float silu_f(float x) {
  return x * (1.0f / (1.0f + __expf(-x)));
}

static __device__ __forceinline__ v8f wmma_bf16(v16bf a, v16bf b, v8f c) {
  return __builtin_amdgcn_wmma_f32_16x16x32_bf16(false, a, false, b, (short)0, c,
                                                 false, false);
}

// ---------------------------------------------------------------------------
// One dense MLP layer over a 128-row tile held in LDS (bf16).
// 8 waves split the N-tiles; each wave keeps 8 accumulator tiles (all M) live
// so every packed weight fragment is loaded from L2 exactly once per block.
// ---------------------------------------------------------------------------
static __device__ __forceinline__ void mlp_layer(const bf16_t* in, int in_stride,
                                                 bf16_t* out, int out_stride,
                                                 const bf16_t* wpk, const float* bias,
                                                 int kTiles, int nTiles, bool act,
                                                 int wave, int lane) {
  for (int n = wave; n < nTiles; n += 8) {
    v8f acc[8] = {};
    for (int k = 0; k < kTiles; ++k) {
      v16bf bf = load_b_frag(wpk, k * nTiles + n, lane);
#pragma unroll
      for (int m = 0; m < 8; ++m) {
        v16bf af = load_a_frag(in, in_stride, m * 16, k * 32, lane);
        acc[m] = wmma_bf16(af, bf, acc[m]);
      }
    }
    const int col = n * 16 + (lane & 15);
    const float bv = bias[col];
#pragma unroll
    for (int m = 0; m < 8; ++m) {
      const int row0 = m * 16 + ((lane >> 4) << 3);
#pragma unroll
      for (int r = 0; r < 8; ++r) {
        float v = acc[m][r] + bv;
        if (act) v = silu_f(v);
        out[(row0 + r) * out_stride + col] = (bf16_t)v;
      }
    }
  }
}

// Final branch layer: scatter-add results to per-node accumulators.
static __device__ __forceinline__ void mlp_layer_scatter(const bf16_t* in, int in_stride,
                                                         float* dst, int dst_stride,
                                                         int dst_cols, const int* colI,
                                                         const bf16_t* wpk,
                                                         const float* bias, int bias_n,
                                                         int kTiles, int nTiles,
                                                         int validRows,
                                                         int wave, int lane) {
  for (int n = wave; n < nTiles; n += 8) {
    v8f acc[8] = {};
    for (int k = 0; k < kTiles; ++k) {
      v16bf bf = load_b_frag(wpk, k * nTiles + n, lane);
#pragma unroll
      for (int m = 0; m < 8; ++m) {
        v16bf af = load_a_frag(in, in_stride, m * 16, k * 32, lane);
        acc[m] = wmma_bf16(af, bf, acc[m]);
      }
    }
    const int col = n * 16 + (lane & 15);
    const float bv = (col < bias_n) ? bias[col] : 0.0f;
    if (col < dst_cols) {
#pragma unroll
      for (int m = 0; m < 8; ++m) {
        const int row0 = m * 16 + ((lane >> 4) << 3);
#pragma unroll
        for (int r = 0; r < 8; ++r) {
          const int row = row0 + r;
          if (row < validRows) {
            const int node = colI[row];
            atomicAdd(dst + (size_t)node * dst_stride + col, acc[m][r] + bv);
          }
        }
      }
    }
  }
}

// ---------------------------------------------------------------------------
// Edge pipeline: 128 edges per block, full message/vector/scalar MLPs.
// ---------------------------------------------------------------------------
#define EPB 128
#define A_STR 360   // holds up to 352-wide activations; 720B rows (16B aligned, staggered banks)
#define B_STR 264
#define M_STR 136
#define EDGE_SMEM (EPB * A_STR * 2 + EPB * B_STR * 2 + EPB * M_STR * 2 + 2 * EPB * 4)
#define NODE_SMEM (2 * 128 * B_STR * 2)

__global__ __launch_bounds__(256) void edge_kernel(
    const int* __restrict__ ei, const float* __restrict__ ea,
    const float* __restrict__ x, const float* __restrict__ h,
    const bf16_t* mW0, const float* mB0, const bf16_t* mW1, const float* mB1,
    const bf16_t* mW2, const float* mB2,
    const bf16_t* vW0, const float* vB0, const bf16_t* vW1, const float* vB1,
    const bf16_t* vW2, const float* vB2,
    const bf16_t* sW0, const float* sB0, const bf16_t* sW1, const float* sB1,
    const bf16_t* sW2, const float* sB2,
    float* outVec, float* agg, int E) {
  extern __shared__ char smem[];
  bf16_t* A = (bf16_t*)smem;                                  // 128 x 360
  bf16_t* B = (bf16_t*)(smem + EPB * A_STR * 2);              // 128 x 264
  bf16_t* M = (bf16_t*)(smem + EPB * (A_STR + B_STR) * 2);    // 128 x 136 (message)
  int* rowI = (int*)(smem + EPB * (A_STR + B_STR + M_STR) * 2);
  int* colI = rowI + EPB;

  const int tid  = threadIdx.x;
  const int lane = tid & 31;
  const int wave = tid >> 5;
  const int e0   = blockIdx.x * EPB;
  const int eCnt = (E - e0 < EPB) ? (E - e0) : EPB;

  // ---- stage edge indices (clamped so padded rows read valid memory) ----
  for (int i = tid; i < EPB; i += 256) {
    int e = e0 + ((i < eCnt) ? i : 0);
    rowI[i] = ei[e];
    colI[i] = ei[E + e];
  }
  __syncthreads();

  // ---- build normalized xij (cols 0..5) ----
  if (tid < EPB) {
    int r = rowI[tid], c = colI[tid];
    float v0 = x[3 * r], v1 = x[3 * r + 1], v2 = x[3 * r + 2];
    float v3 = x[3 * c], v4 = x[3 * c + 1], v5 = x[3 * c + 2];
    float nrm = sqrtf(v0 * v0 + v1 * v1 + v2 * v2 + v3 * v3 + v4 * v4 + v5 * v5);
    float inv = 1.0f / fmaxf(nrm, 1e-12f);
    bf16_t* a = A + tid * A_STR;
    a[0] = (bf16_t)(v0 * inv); a[1] = (bf16_t)(v1 * inv); a[2] = (bf16_t)(v2 * inv);
    a[3] = (bf16_t)(v3 * inv); a[4] = (bf16_t)(v4 * inv); a[5] = (bf16_t)(v5 * inv);
  }
  // zero K-padding cols 326..351
  for (int i = tid; i < EPB * 26; i += 256) {
    A[(i / 26) * A_STR + 326 + (i % 26)] = (bf16_t)0.0f;
  }
  // h[row] -> cols 6..133
  for (int i = tid; i < EPB * 128; i += 256) {
    int e = i >> 7, c = i & 127;
    A[e * A_STR + 6 + c] = (bf16_t)h[(size_t)rowI[e] * 128 + c];
  }
  // h[col] -> cols 134..261
  for (int i = tid; i < EPB * 128; i += 256) {
    int e = i >> 7, c = i & 127;
    A[e * A_STR + 134 + c] = (bf16_t)h[(size_t)colI[e] * 128 + c];
  }
  // edge_attr -> cols 262..325
  for (int i = tid; i < EPB * 64; i += 256) {
    int e = i >> 6, c = i & 63;
    int ge = e0 + ((e < eCnt) ? e : 0);
    A[e * A_STR + 262 + c] = (bf16_t)ea[(size_t)ge * 64 + c];
  }
  __syncthreads();

  // ---- message MLP: 326(352) -> 256 -> 256 -> 128 ----
  mlp_layer(A, A_STR, B, B_STR, mW0, mB0, 11, 16, true, wave, lane);
  __syncthreads();
  mlp_layer(B, B_STR, A, A_STR, mW1, mB1, 8, 16, true, wave, lane);
  __syncthreads();
  mlp_layer(A, A_STR, M, M_STR, mW2, mB2, 8, 8, false, wave, lane);
  __syncthreads();

  // ---- vector branch: 128 -> 256 -> 256 -> 3, scatter into outVec ----
  mlp_layer(M, M_STR, B, B_STR, vW0, vB0, 4, 16, true, wave, lane);
  __syncthreads();
  mlp_layer(B, B_STR, A, A_STR, vW1, vB1, 8, 16, true, wave, lane);
  __syncthreads();
  mlp_layer_scatter(A, A_STR, outVec, 3, 3, colI, vW2, vB2, 3, 8, 1, eCnt, wave, lane);
  __syncthreads();

  // ---- scalar branch: 128 -> 256 -> 256 -> 128, scatter into agg ----
  mlp_layer(M, M_STR, B, B_STR, sW0, sB0, 4, 16, true, wave, lane);
  __syncthreads();
  mlp_layer(B, B_STR, A, A_STR, sW1, sB1, 8, 16, true, wave, lane);
  __syncthreads();
  mlp_layer_scatter(A, A_STR, agg, 128, 128, colI, sW2, sB2, 128, 8, 8, eCnt, wave, lane);
}

// ---------------------------------------------------------------------------
// Node MLP: agg(128) -> 256 -> 256 -> 128, +h residual.
// ---------------------------------------------------------------------------
__global__ __launch_bounds__(256) void node_kernel(
    const float* __restrict__ agg, const float* __restrict__ h,
    const bf16_t* W0, const float* B0, const bf16_t* W1, const float* B1,
    const bf16_t* W2, const float* B2, float* outNode, int N) {
  extern __shared__ char smem[];
  bf16_t* A = (bf16_t*)smem;                       // 128 x 264
  bf16_t* B = (bf16_t*)(smem + 128 * B_STR * 2);   // 128 x 264

  const int tid  = threadIdx.x;
  const int lane = tid & 31;
  const int wave = tid >> 5;
  const int r0   = blockIdx.x * 128;

  for (int i = tid; i < 128 * 128; i += 256) {
    int e = i >> 7, c = i & 127;
    int row = r0 + e;
    float v = (row < N) ? agg[(size_t)row * 128 + c] : 0.0f;
    A[e * B_STR + c] = (bf16_t)v;
  }
  __syncthreads();
  mlp_layer(A, B_STR, B, B_STR, W0, B0, 4, 16, true, wave, lane);
  __syncthreads();
  mlp_layer(B, B_STR, A, B_STR, W1, B1, 8, 16, true, wave, lane);
  __syncthreads();
  // final layer fused with bias + h residual, direct f32 store
  for (int n = wave; n < 8; n += 8) {
    v8f acc[8] = {};
    for (int k = 0; k < 8; ++k) {
      v16bf bf = load_b_frag(W2, k * 8 + n, lane);
#pragma unroll
      for (int m = 0; m < 8; ++m) {
        v16bf af = load_a_frag(A, B_STR, m * 16, k * 32, lane);
        acc[m] = wmma_bf16(af, bf, acc[m]);
      }
    }
    const int col = n * 16 + (lane & 15);
    const float bv = B2[col];
#pragma unroll
    for (int m = 0; m < 8; ++m) {
      const int row0 = r0 + m * 16 + ((lane >> 4) << 3);
#pragma unroll
      for (int r = 0; r < 8; ++r) {
        int row = row0 + r;
        if (row < N)
          outNode[(size_t)row * 128 + col] = acc[m][r] + bv + h[(size_t)row * 128 + col];
      }
    }
  }
}

// ---------------------------------------------------------------------------
// Weight pre-pack: f32 row-major (K x N) -> bf16 WMMA B-fragments, zero padded.
// ---------------------------------------------------------------------------
__global__ void pack_kernel(const float* __restrict__ W, bf16_t* __restrict__ dst,
                            int K, int N, int kTiles, int nTiles) {
  int total = kTiles * nTiles * 512;
  for (int idx = blockIdx.x * blockDim.x + threadIdx.x; idx < total;
       idx += gridDim.x * blockDim.x) {
    int tile = idx >> 9;
    int rem  = idx & 511;
    int lane = rem >> 4;
    int e    = rem & 15;
    int kt   = tile / nTiles, nt = tile % nTiles;
    int k = kt * 32 + ((e >> 3) << 4) + ((lane >> 4) << 3) + (e & 7);   // B layout: elem e -> K = 16*(e/8) + 8*laneHi + e%8
    int n = nt * 16 + (lane & 15);
    float v = (k < K && n < N) ? W[(size_t)k * N + n] : 0.0f;
    dst[idx] = (bf16_t)v;
  }
}

__global__ void init_kernel(float* outVec, const float* x, float* agg,
                            int nVec, int nAgg) {
  int stride = gridDim.x * blockDim.x;
  for (int i = blockIdx.x * blockDim.x + threadIdx.x; i < nAgg; i += stride)
    agg[i] = 0.0f;
  for (int i = blockIdx.x * blockDim.x + threadIdx.x; i < nVec; i += stride)
    outVec[i] = x[i];
}

// ---------------------------------------------------------------------------
extern "C" void kernel_launch(void* const* d_in, const int* in_sizes, int n_in,
                              void* d_out, int out_size, void* d_ws, size_t ws_size,
                              hipStream_t stream) {
  const int E = in_sizes[24] / 2;
  const int N = in_sizes[27] / 128;

  const int*   ei = (const int*)d_in[24];
  const float* ea = (const float*)d_in[25];
  const float* x  = (const float*)d_in[26];
  const float* h  = (const float*)d_in[27];

  float* outVec  = (float*)d_out;            // (N,3)
  float* outNode = outVec + (size_t)N * 3;   // (N,128)

  char*  ws  = (char*)d_ws;
  float* agg = (float*)ws;                   // (N,128) f32 scatter accumulator
  bf16_t* pk = (bf16_t*)(ws + (size_t)N * 128 * sizeof(float));

  // 12 weight matrices: {input index, K, N, kTiles, nTiles}
  const int meta[12][5] = {
      {0, 326, 256, 11, 16}, {2, 256, 256, 8, 16}, {4, 256, 128, 8, 8},   // message
      {6, 128, 256, 4, 16},  {8, 256, 256, 8, 16}, {10, 256, 3, 8, 1},    // vector
      {12, 128, 256, 4, 16}, {14, 256, 256, 8, 16}, {16, 256, 128, 8, 8}, // scalar
      {18, 128, 256, 4, 16}, {20, 256, 256, 8, 16}, {22, 256, 128, 8, 8}, // node
  };
  bf16_t* wp[12];
  size_t off = 0;
  for (int i = 0; i < 12; ++i) {
    wp[i] = pk + off;
    int elems = meta[i][3] * meta[i][4] * 512;
    off += (size_t)elems;
    pack_kernel<<<(elems + 255) / 256, 256, 0, stream>>>(
        (const float*)d_in[meta[i][0]], wp[i], meta[i][1], meta[i][2],
        meta[i][3], meta[i][4]);
  }

  {
    int nAgg = N * 128, nVec = N * 3;
    init_kernel<<<1024, 256, 0, stream>>>(outVec, x, agg, nVec, nAgg);
  }

  // Idempotent, not a stream op; called every time (no static guards allowed).
  hipFuncSetAttribute((const void*)edge_kernel,
                      hipFuncAttributeMaxDynamicSharedMemorySize, EDGE_SMEM);
  hipFuncSetAttribute((const void*)node_kernel,
                      hipFuncAttributeMaxDynamicSharedMemorySize, NODE_SMEM);

  edge_kernel<<<(E + EPB - 1) / EPB, 256, EDGE_SMEM, stream>>>(
      ei, ea, x, h,
      wp[0], (const float*)d_in[1], wp[1], (const float*)d_in[3],
      wp[2], (const float*)d_in[5],
      wp[3], (const float*)d_in[7], wp[4], (const float*)d_in[9],
      wp[5], (const float*)d_in[11],
      wp[6], (const float*)d_in[13], wp[7], (const float*)d_in[15],
      wp[8], (const float*)d_in[17],
      outVec, agg, E);

  node_kernel<<<(N + 127) / 128, 256, NODE_SMEM, stream>>>(
      agg, h,
      wp[9], (const float*)d_in[19], wp[10], (const float*)d_in[21],
      wp[11], (const float*)d_in[23], outNode, N);
}
